// DKVMN_QUE_75050258530304
// MI455X (gfx1250) — compile-verified, matrix operand order checked by reading, not run on hardware
//
#include <hip/hip_runtime.h>
#include <hip/hip_bf16.h>
#include <cstdint>

typedef __attribute__((ext_vector_type(16))) __bf16 bf16x16;
typedef __attribute__((ext_vector_type(8)))  __bf16 bf16x8;
typedef __attribute__((ext_vector_type(8)))  float  f32x8;
typedef __attribute__((ext_vector_type(4)))  int    i32x4;

#define NQ    10000
#define BB    64
#define LL    200
#define DKD   128
#define NSLOT 64
#define ROWS  (BB * LL)   /* 12800 */

#if __has_builtin(__builtin_amdgcn_global_load_async_to_lds_b128) && \
    __has_builtin(__builtin_amdgcn_s_wait_asynccnt)
#define HAS_ASYNC_LDS 1
#define AS1 __attribute__((address_space(1)))
#define AS3 __attribute__((address_space(3)))
// Builtin signature (from hipcc diagnostic): (int4 __device__*, int4 LDS*, imm, imm)
#define ASYNC_LD_B128(gsrc, ldst) \
    __builtin_amdgcn_global_load_async_to_lds_b128( \
        (AS1 i32x4*)(AS1 void*)(gsrc), (AS3 i32x4*)(AS3 void*)(ldst), 0, 0)
#endif

__device__ __forceinline__ uint16_t f2bf(float f) {
    uint32_t u = __builtin_bit_cast(uint32_t, f);
    u += 0x7FFFu + ((u >> 16) & 1u);           // round-to-nearest-even
    return (uint16_t)(u >> 16);
}

__device__ __forceinline__ float sigmoidf(float x) {
    return 1.0f / (1.0f + __expf(-x));
}

// Load one lane's slice of a 16x32 (MxK) bf16 WMMA operand.
// Lane groups: lanes 0-15 hold K {kb+0..7, kb+16..23}; lanes 16-31 hold {kb+8..15, kb+24..31}.
__device__ __forceinline__ bf16x16 load_k32(const uint16_t* rowbase, int kb, int lane) {
    const uint16_t* p = rowbase + kb + ((lane >> 4) << 3);
    bf16x8 lo = *(const bf16x8*)(p);
    bf16x8 hi = *(const bf16x8*)(p + 16);
    bf16x16 r;
#pragma unroll
    for (int i = 0; i < 8; ++i) { r[i] = lo[i]; r[i + 8] = hi[i]; }
    return r;
}

// ---------------------------------------------------------------------------
// Kernel 1: f32 -> bf16 weight conversion
// ---------------------------------------------------------------------------
__global__ void convert_bf16_kernel(const float* __restrict__ src,
                                    uint16_t* __restrict__ dst, int n) {
    int i = blockIdx.x * blockDim.x + threadIdx.x;
    if (i < n) dst[i] = f2bf(src[i]);
}

// ---------------------------------------------------------------------------
// Kernel 2: embedding gather + bf16 convert. One block per row, 128 threads.
// ---------------------------------------------------------------------------
__global__ void __launch_bounds__(DKD)
gather_kernel(const int* __restrict__ qs, const int* __restrict__ cs,
              const float* __restrict__ q_emb, const float* __restrict__ v_emb,
              uint16_t* __restrict__ kbf, uint16_t* __restrict__ vbf) {
    int row = blockIdx.x;
    int d   = threadIdx.x;
    int q   = qs[row];
    int x   = q + NQ * cs[row];
    kbf[row * DKD + d] = f2bf(q_emb[q * DKD + d]);
    vbf[row * DKD + d] = f2bf(v_emb[x * DKD + d]);
}

// ---------------------------------------------------------------------------
// Kernel 3: fused  w = softmax(k Mk^T),  e = sigmoid(v eW^T + eb),
//                  a = tanh(v aW^T + ab).
// One block per 16-row tile (800 blocks), 8 waves; 20 16x16 column tiles.
// ---------------------------------------------------------------------------
__global__ void __launch_bounds__(256)
wea_kernel(const uint16_t* __restrict__ kbf, const uint16_t* __restrict__ vbf,
           const uint16_t* __restrict__ Mkbf, const uint16_t* __restrict__ eWbf,
           const uint16_t* __restrict__ aWbf,
           const float* __restrict__ eb, const float* __restrict__ ab,
           float* __restrict__ w, float* __restrict__ e, float* __restrict__ a) {
    __shared__ float logits[16 * NSLOT];

    const int row0 = blockIdx.x * 16;
    const int lane = threadIdx.x & 31;
    const int wv   = threadIdx.x >> 5;
    const int m    = lane & 15;

    const uint16_t* krow = kbf + (size_t)(row0 + m) * DKD;
    const uint16_t* vrow = vbf + (size_t)(row0 + m) * DKD;

    bf16x16 ak[4], av[4];
#pragma unroll
    for (int i = 0; i < 4; ++i) {
        ak[i] = load_k32(krow, i * 32, lane);
        av[i] = load_k32(vrow, i * 32, lane);
    }

    for (int tile = wv; tile < 20; tile += 8) {
        const uint16_t* W;
        int n0;
        bool useK = false;
        if (tile < 4)        { W = Mkbf; n0 = tile * 16;        useK = true; }
        else if (tile < 12)  { W = eWbf; n0 = (tile - 4) * 16;  }
        else                 { W = aWbf; n0 = (tile - 12) * 16; }

        const uint16_t* wrow = W + (size_t)(n0 + m) * DKD;

        // Preload all 4 B-tiles first -> one load clause, one wait, then
        // 4 back-to-back WMMAs instead of load/wait/wmma serialization.
        bf16x16 bmat[4];
#pragma unroll
        for (int i = 0; i < 4; ++i) bmat[i] = load_k32(wrow, i * 32, lane);

        f32x8 acc = {};
#pragma unroll
        for (int i = 0; i < 4; ++i) {
            bf16x16 amat = useK ? ak[i] : av[i];
            acc = __builtin_amdgcn_wmma_f32_16x16x32_bf16(
                false, amat, false, bmat[i], (short)0, acc, false, false);
        }

        const int rb = (lane >> 4) * 8;   // row base within tile
        const int n  = lane & 15;         // column within tile
        if (tile < 4) {
#pragma unroll
            for (int i = 0; i < 8; ++i)
                logits[(rb + i) * NSLOT + n0 + n] = acc[i];
        } else if (tile < 12) {
            float bias = eb[n0 + n];
#pragma unroll
            for (int i = 0; i < 8; ++i)
                e[(size_t)(row0 + rb + i) * DKD + n0 + n] = sigmoidf(acc[i] + bias);
        } else {
            float bias = ab[n0 + n];
#pragma unroll
            for (int i = 0; i < 8; ++i)
                a[(size_t)(row0 + rb + i) * DKD + n0 + n] = tanhf(acc[i] + bias);
        }
    }
    __syncthreads();

    // Row-wise softmax over the 64 slots.
    if (threadIdx.x < 16) {
        int mm = threadIdx.x;
        float mx = -3.0e38f;
        for (int s = 0; s < NSLOT; ++s) mx = fmaxf(mx, logits[mm * NSLOT + s]);
        float sum = 0.0f;
        for (int s = 0; s < NSLOT; ++s) sum += __expf(logits[mm * NSLOT + s] - mx);
        float inv = 1.0f / sum;
        for (int s = 0; s < NSLOT; ++s)
            w[(size_t)(row0 + mm) * NSLOT + s] = __expf(logits[mm * NSLOT + s] - mx) * inv;
    }
}

// ---------------------------------------------------------------------------
// Kernel 4: sequential memory scan. One block per batch element (64 blocks),
// 128 threads; Mv[64][128] resident in LDS (32 KB). Thread d owns column d.
// Per-step w/e/a inputs (1.25 KB) stream in via async global->LDS copies
// (ASYNCcnt) into a depth-3 ring buffer, hiding global latency behind the
// 64-iteration rank-1 update of earlier steps.
// Ring slot layout (320 floats): [0..63]=w, [64..191]=e, [192..319]=a.
// ---------------------------------------------------------------------------
#define SCAN_DEPTH 3
#define STAGEF     (NSLOT + 2 * DKD)   /* 320 floats per stage */
#define NCHUNK     (STAGEF / 4)        /* 80 x 16B chunks      */

__global__ void __launch_bounds__(DKD)
scan_kernel(const float* __restrict__ w, const float* __restrict__ e,
            const float* __restrict__ a, const float* __restrict__ Mv0,
            uint16_t* __restrict__ readsbf) {
    __shared__ float Mv[NSLOT * DKD];
    __shared__ __attribute__((aligned(16))) float ring[SCAN_DEPTH][STAGEF];

    const int b   = blockIdx.x;
    const int tid = threadIdx.x;

    for (int i = tid; i < NSLOT * DKD; i += DKD) Mv[i] = Mv0[i];

#ifdef HAS_ASYNC_LDS
    // Per-thread chunk mapping (threads 0..79 move one 16B chunk per stage).
    const float* gbase = nullptr;
    int goffs = 0;
    if (tid < 16)          { gbase = w; goffs = tid * 4;        }
    else if (tid < 48)     { gbase = e; goffs = (tid - 16) * 4; }
    else if (tid < NCHUNK) { gbase = a; goffs = (tid - 48) * 4; }

    // Prologue: issue stages 0..SCAN_DEPTH-2.
    for (int t = 0; t < SCAN_DEPTH - 1; ++t) {
        if (tid < NCHUNK) {
            const size_t row = (size_t)b * LL + t;
            const float* gsrc = (gbase == w) ? gbase + row * NSLOT + goffs
                                             : gbase + row * DKD + goffs;
            ASYNC_LD_B128(gsrc, &ring[t][tid * 4]);
        }
    }
    __syncthreads();

    for (int t = 0; t < LL; ++t) {
        // Issue stage t+DEPTH-1 (its slot was fully consumed at step t-1).
        const int tp = t + SCAN_DEPTH - 1;
        if (tp < LL && tid < NCHUNK) {
            const size_t prow = (size_t)b * LL + tp;
            const float* gsrc = (gbase == w) ? gbase + prow * NSLOT + goffs
                                             : gbase + prow * DKD + goffs;
            ASYNC_LD_B128(gsrc, &ring[tp % SCAN_DEPTH][tid * 4]);
        }
        __builtin_amdgcn_s_wait_asynccnt(SCAN_DEPTH - 1);  // stage t landed
        __syncthreads();

        const float* st = ring[t % SCAN_DEPTH];
        const float ed = st[NSLOT + tid];
        const float ad = st[NSLOT + DKD + tid];
        float r = 0.0f;
#pragma unroll 8
        for (int s = 0; s < NSLOT; ++s) {
            float ws = st[s];
            float mv = Mv[s * DKD + tid];
            r = fmaf(ws, mv, r);                                  // read (pre-update)
            Mv[s * DKD + tid] = mv * (1.0f - ws * ed) + ws * ad;  // rank-1 update
        }
        readsbf[((size_t)b * LL + t) * DKD + tid] = f2bf(r);
        __syncthreads();   // slot t%DEPTH free for reuse at step t+1's issue
    }
#else
    // Fallback: synchronous staging through LDS slot 0.
    __syncthreads();
    for (int t = 0; t < LL; ++t) {
        const size_t row = (size_t)b * LL + t;
        if (tid < NSLOT) ring[0][tid] = w[row * NSLOT + tid];
        ring[0][NSLOT + tid]       = e[row * DKD + tid];
        ring[0][NSLOT + DKD + tid] = a[row * DKD + tid];
        __syncthreads();
        const float ed = ring[0][NSLOT + tid];
        const float ad = ring[0][NSLOT + DKD + tid];
        float r = 0.0f;
#pragma unroll 8
        for (int s = 0; s < NSLOT; ++s) {
            float ws = ring[0][s];
            float mv = Mv[s * DKD + tid];
            r = fmaf(ws, mv, r);
            Mv[s * DKD + tid] = mv * (1.0f - ws * ed) + ws * ad;
        }
        readsbf[row * DKD + tid] = f2bf(r);
        __syncthreads();
    }
#endif
}

// ---------------------------------------------------------------------------
// Kernel 5: f = tanh([reads | k] fW^T + fb)  then  out = sigmoid(f . pW + pb).
// One block per 16-row tile, 8 waves; wave i owns column tile n0 = 16*i, K=256.
// ---------------------------------------------------------------------------
__global__ void __launch_bounds__(256)
fp_kernel(const uint16_t* __restrict__ readsbf, const uint16_t* __restrict__ kbf,
          const uint16_t* __restrict__ fWbf, const float* __restrict__ fb,
          const float* __restrict__ pW, const float* __restrict__ pb,
          float* __restrict__ out) {
    __shared__ float ftile[16 * DKD];

    const int row0 = blockIdx.x * 16;
    const int lane = threadIdx.x & 31;
    const int wv   = threadIdx.x >> 5;
    const int m    = lane & 15;
    const int n0   = wv * 16;

    const uint16_t* rrow = readsbf + (size_t)(row0 + m) * DKD;
    const uint16_t* krow = kbf     + (size_t)(row0 + m) * DKD;
    const uint16_t* wrow = fWbf    + (size_t)(n0 + m) * (2 * DKD);

    // Preload all operands -> big load clause, then 8 chained WMMAs.
    bf16x16 amat[8], bmat[8];
#pragma unroll
    for (int kb = 0; kb < 8; ++kb) {
        const int kk = kb * 32;
        amat[kb] = (kk < DKD) ? load_k32(rrow, kk, lane)
                              : load_k32(krow, kk - DKD, lane);
        bmat[kb] = load_k32(wrow, kk, lane);
    }
    f32x8 acc = {};
#pragma unroll
    for (int kb = 0; kb < 8; ++kb)
        acc = __builtin_amdgcn_wmma_f32_16x16x32_bf16(
            false, amat[kb], false, bmat[kb], (short)0, acc, false, false);

    const int rb = (lane >> 4) * 8;
    const int n  = lane & 15;
    const float bias = fb[n0 + n];
#pragma unroll
    for (int i = 0; i < 8; ++i)
        ftile[(rb + i) * DKD + n0 + n] = tanhf(acc[i] + bias);
    __syncthreads();

    if (threadIdx.x < 16) {
        int mm = threadIdx.x;
        float s = pb[0];
        for (int d = 0; d < DKD; ++d) s = fmaf(ftile[mm * DKD + d], pW[d], s);
        out[row0 + mm] = sigmoidf(s);
    }
}

// ---------------------------------------------------------------------------
extern "C" void kernel_launch(void* const* d_in, const int* in_sizes, int n_in,
                              void* d_out, int out_size, void* d_ws, size_t ws_size,
                              hipStream_t stream) {
    const int*   question = (const int*)d_in[0];
    const int*   correct  = (const int*)d_in[1];
    const float* q_emb    = (const float*)d_in[2];
    const float* v_emb    = (const float*)d_in[3];
    const float* Mk       = (const float*)d_in[4];
    const float* Mv0      = (const float*)d_in[5];
    const float* eW       = (const float*)d_in[6];
    const float* eb       = (const float*)d_in[7];
    const float* aW       = (const float*)d_in[8];
    const float* ab       = (const float*)d_in[9];
    const float* fW       = (const float*)d_in[10];
    const float* fb       = (const float*)d_in[11];
    const float* pW       = (const float*)d_in[12];
    const float* pb       = (const float*)d_in[13];
    float* out = (float*)d_out;

    // Workspace carve-up (256B aligned slices).
    size_t off = 0;
    char* base = (char*)d_ws;
    auto carve = [&](size_t bytes) -> char* {
        char* p = base + off;
        off += (bytes + 255) & ~(size_t)255;
        return p;
    };
    uint16_t* kbf     = (uint16_t*)carve((size_t)ROWS * DKD * 2);
    uint16_t* vbf     = (uint16_t*)carve((size_t)ROWS * DKD * 2);
    uint16_t* readsbf = (uint16_t*)carve((size_t)ROWS * DKD * 2);
    float*    wbuf    = (float*)   carve((size_t)ROWS * NSLOT * 4);
    float*    ebuf    = (float*)   carve((size_t)ROWS * DKD * 4);
    float*    abuf    = (float*)   carve((size_t)ROWS * DKD * 4);
    uint16_t* Mkbf    = (uint16_t*)carve((size_t)NSLOT * DKD * 2);
    uint16_t* eWbf    = (uint16_t*)carve((size_t)DKD * DKD * 2);
    uint16_t* aWbf    = (uint16_t*)carve((size_t)DKD * DKD * 2);
    uint16_t* fWbf    = (uint16_t*)carve((size_t)DKD * 2 * DKD * 2);

    // 1. Weight conversion to bf16.
    convert_bf16_kernel<<<(NSLOT * DKD + 255) / 256, 256, 0, stream>>>(Mk, Mkbf, NSLOT * DKD);
    convert_bf16_kernel<<<(DKD * DKD + 255) / 256, 256, 0, stream>>>(eW, eWbf, DKD * DKD);
    convert_bf16_kernel<<<(DKD * DKD + 255) / 256, 256, 0, stream>>>(aW, aWbf, DKD * DKD);
    convert_bf16_kernel<<<(DKD * 2 * DKD + 255) / 256, 256, 0, stream>>>(fW, fWbf, DKD * 2 * DKD);

    // 2. Embedding gathers.
    gather_kernel<<<ROWS, DKD, 0, stream>>>(question, correct, q_emb, v_emb, kbf, vbf);

    // 3. Fused w / e / a (WMMA bf16 GEMMs + softmax/sigmoid/tanh).
    wea_kernel<<<ROWS / 16, 256, 0, stream>>>(kbf, vbf, Mkbf, eWbf, aWbf, eb, ab,
                                              wbuf, ebuf, abuf);

    // 4. Sequential DKVMN memory scan with async-LDS prefetch ring.
    scan_kernel<<<BB, DKD, 0, stream>>>(wbuf, ebuf, abuf, Mv0, readsbf);

    // 5. Output head: f-GEMM (K=256, WMMA) + tanh + pW dot + sigmoid.
    fp_kernel<<<ROWS / 16, 256, 0, stream>>>(readsbf, kbf, fWbf, fb, pW, pb, out);
}
